// PFN_24781961298185
// MI455X (gfx1250) — matrix operand order repaired and checked by reading, not compiled, not run on hardware
//
#include <hip/hip_runtime.h>

// ---------------- problem constants ----------------
#define LL   100
#define BB   8
#define DD   768
#define HH   300
#define HP   320        // H padded to multiple of 32 (WMMA K granularity)
#define G5   1500       // 5*H
#define GP   1504       // padded gate row in LDS
#define NTT  8
#define NRR  12
#define MROWS (LL*BB)   // 800

typedef __bf16 bf16;
typedef bf16  v16bf __attribute__((ext_vector_type(16)));
typedef bf16  v8bf  __attribute__((ext_vector_type(8)));
typedef float v8f   __attribute__((ext_vector_type(8)));

// ---------------- WMMA fragment helpers ----------------
__device__ __forceinline__ v16bf frag_combine(v8bf lo, v8bf hi) {
  v16bf r;
#pragma unroll
  for (int e = 0; e < 8; ++e) { r[e] = lo[e]; r[e + 8] = hi[e]; }
  return r;
}
// A (16x32 bf16): lane = m(0..15) + 16*khalf ; 16 consecutive K from row[k0 + khalf*16]
__device__ __forceinline__ v16bf load_afrag(const bf16* rowPtr, int k0, int khalf) {
  v8bf lo = *reinterpret_cast<const v8bf*>(rowPtr + k0 + khalf * 16);
  v8bf hi = *reinterpret_cast<const v8bf*>(rowPtr + k0 + khalf * 16 + 8);
  return frag_combine(lo, hi);
}
// B (32x16 bf16): lane = n(0..15) + 16*khalf ; 16 consecutive K from W[n][k0 + khalf*16]
__device__ __forceinline__ v16bf load_bfrag(const bf16* p) {
  v8bf lo = *reinterpret_cast<const v8bf*>(p);
  v8bf hi = *reinterpret_cast<const v8bf*>(p + 8);
  return frag_combine(lo, hi);
}
__device__ __forceinline__ v8f wmma_bf16(v16bf a, v16bf b, v8f c) {
  return __builtin_amdgcn_wmma_f32_16x16x32_bf16(false, a, false, b, (short)0, c, false, false);
}

// ---------------- utility kernels ----------------
// f32 -> bf16 with per-segment K padding: src [N][nseg*Ksrc] -> dst [N][nseg*Kdst] (pad = 0)
__global__ void cvt_pad_kernel(const float* __restrict__ src, bf16* __restrict__ dst,
                               int nrows, int Ksrc, int Kdst, int nseg) {
  long total = (long)nrows * nseg * Kdst;
  for (long idx = (long)blockIdx.x * blockDim.x + threadIdx.x; idx < total;
       idx += (long)gridDim.x * blockDim.x) {
    int k = (int)(idx % Kdst);
    long r = idx / Kdst;
    int s = (int)(r % nseg);
    long n = r / nseg;
    float v = (k < Ksrc) ? src[(n * nseg + s) * Ksrc + k] : 0.f;
    dst[(n * nseg + s) * Kdst + k] = (bf16)v;
  }
}

__global__ void zero_bf16_kernel(bf16* __restrict__ p, long n) {
  for (long i = (long)blockIdx.x * blockDim.x + threadIdx.x; i < n;
       i += (long)gridDim.x * blockDim.x)
    p[i] = (bf16)0.f;
}

// column max over L: pre [L*B][H] f32 -> out [B][HP] bf16 (pad cols = 0)
__global__ void colmax_kernel(const float* __restrict__ pre, bf16* __restrict__ outBf) {
  int idx = blockIdx.x * blockDim.x + threadIdx.x;
  if (idx >= BB * HP) return;
  int b = idx / HP, n = idx % HP;
  float m = 0.f;
  if (n < HH) {
    m = -3.0e38f;
    for (int l = 0; l < LL; ++l) m = fmaxf(m, pre[((size_t)l * BB + b) * HH + n]);
  }
  outBf[idx] = (bf16)m;
}

// ---------------- generic bf16 WMMA GEMM (16x64 strip per wave) ----------------
// C[M,N] = act( A[M,K] @ B^T + bias ),  B row-major [N][ldb] (i.e. B[k][n] = Bw[n][k])
// A may be the concat of two matrices split at kSplit (kSplit multiple of 32).
// Out-of-range A rows / B cols are CLAMPED on load and the edge strip always issues
// all NSUB WMMAs (their D rows/cols are never stored, and WMMA rows/cols are
// independent), so the inner loop is fully straight-line: no predication, no phis.
#define NSUB 4
__global__ void gemm_bf16_kernel(const bf16* __restrict__ A1, int lda1,
                                 const bf16* __restrict__ A2, int lda2, int kSplit,
                                 const bf16* __restrict__ Bw, int ldb,
                                 const float* __restrict__ bias,
                                 float* __restrict__ Cout, int ldc,
                                 int M, int N, int K, int act) {
  int wavesPerBlk = blockDim.x >> 5;
  int wv = blockIdx.x * wavesPerBlk + (threadIdx.x >> 5);
  int Mt = (M + 15) >> 4, Nt = (N + 15) >> 4;
  int NtS = (Nt + NSUB - 1) / NSUB;
  if (wv >= Mt * NtS) return;
  int tm = wv / NtS, tn0 = (wv % NtS) * NSUB;
  int lane = threadIdx.x & 31;
  int ml = lane & 15, khalf = lane >> 4;
  int row = tm * 16 + ml;
  int rowc = (row < M) ? row : (M - 1);       // clamp (discarded rows)
  const bf16* Arow1 = A1 + (size_t)rowc * lda1;
  const bf16* Arow2 = A2 + (size_t)rowc * lda2;
  const bf16* Brow[NSUB];
#pragma unroll
  for (int s = 0; s < NSUB; ++s) {
    int coll = (tn0 + s) * 16 + ml;
    int colc = (coll < N) ? coll : (N - 1);   // clamp (discarded cols)
    Brow[s] = Bw + (size_t)colc * ldb + khalf * 16;
  }

  v8f acc[NSUB] = {};
  int kEnd1 = (kSplit < K) ? kSplit : K;
  for (int k0 = 0; k0 < kEnd1; k0 += 32) {
    v16bf af = load_afrag(Arow1, k0, khalf);
    __builtin_prefetch(Brow[0] + k0 + 32, 0, 3);       // -> global_prefetch_b8
#pragma unroll
    for (int s = 0; s < NSUB; ++s)
      acc[s] = wmma_bf16(af, load_bfrag(Brow[s] + k0), acc[s]);
  }
  for (int k0 = kEnd1; k0 < K; k0 += 32) {
    v16bf af = load_afrag(Arow2, k0 - kEnd1, khalf);
    __builtin_prefetch(Brow[0] + k0 + 32, 0, 3);
#pragma unroll
    for (int s = 0; s < NSUB; ++s)
      acc[s] = wmma_bf16(af, load_bfrag(Brow[s] + k0), acc[s]);
  }

#pragma unroll
  for (int s = 0; s < NSUB; ++s) {
    int ccol = (tn0 + s) * 16 + (lane & 15);
    if (ccol < N) {
      float bv = bias ? bias[ccol] : 0.f;
#pragma unroll
      for (int r = 0; r < 8; ++r) {
        int rr = tm * 16 + ((lane < 16) ? r : r + 8);
        if (rr < M) {
          float v = acc[s][r] + bv;
          if (act == 1) v = tanhf(v);
          Cout[(size_t)rr * ldc + ccol] = v;
        }
      }
    }
  }
}

// ---------------- recurrent scan: one resident workgroup ----------------
// LDS layout (62,976 B):
//   [0      .. 48128) : sG  f32 [8][1504]  gates  (aliased later by sCC bf16 [8][960])
//   [48128  .. 57856) : sC  f32 [8][304]   cell state
//   [57856  .. 62976) : sH  bf16 [8][320]  hidden state (K-padded, pad stays 0)
__global__ __launch_bounds__(1024) void scan_kernel(
    const float* __restrict__ xg, const float* __restrict__ bh, const float* __restrict__ bt,
    const bf16* __restrict__ WhBf,   // [1500][320]
    const bf16* __restrict__ WtBf,   // [300][960]
    bf16* __restrict__ hNer, bf16* __restrict__ hRe, bf16* __restrict__ hShare) {
  __shared__ __align__(16) unsigned char smem[62976];
  float* sG  = reinterpret_cast<float*>(smem);
  bf16*  sCC = reinterpret_cast<bf16*>(smem);
  float* sC  = reinterpret_cast<float*>(smem + 48128);
  bf16*  sH  = reinterpret_cast<bf16*>(smem + 48128 + 9728);

  const int tid = threadIdx.x, wv = tid >> 5, lane = tid & 31;
  const int ml = lane & 15, khalf = lane >> 4;
  const int mc = (ml < BB) ? ml : (BB - 1);   // clamp A rows 8..15 (D rows discarded)

  for (int i = tid; i < BB * HP; i += 1024) sH[i] = (bf16)0.f;
  for (int i = tid; i < BB * 304; i += 1024) sC[i] = 0.f;
  __syncthreads();

  for (int t = 0; t < LL; ++t) {
    // ---- phase 1: gates = xg[t] + h @ Wh^T + bh  -> sG [8][1500]
    // strip of 3 tiles per wave: one LDS A-fragment feeds 3 WMMAs per k-chunk.
    // All 3 WMMAs are unconditional (B pointers clamped; stores guarded by n<G5).
    {
      int tile0 = wv * 3;
      v8f acc[3] = {};
      const bf16* Bp[3];
#pragma unroll
      for (int s = 0; s < 3; ++s) {
        int n = (tile0 + s) * 16 + ml;
        int nc = (n < G5) ? n : (G5 - 1);
        Bp[s] = WhBf + (size_t)nc * HP + khalf * 16;
      }
      const bf16* Arow = sH + (size_t)mc * HP;
#pragma unroll
      for (int kc = 0; kc < 10; ++kc) {
        int k0 = kc * 32;
        v16bf af = load_afrag(Arow, k0, khalf);
#pragma unroll
        for (int s = 0; s < 3; ++s)
          acc[s] = wmma_bf16(af, load_bfrag(Bp[s] + k0), acc[s]);
      }
      if (lane < 16) {
        const float* xr = xg + (size_t)t * BB * G5;
#pragma unroll
        for (int s = 0; s < 3; ++s) {
          int n = (tile0 + s) * 16 + ml;
          if (n < G5) {
#pragma unroll
            for (int r = 0; r < 8; ++r)
              sG[r * GP + n] = acc[s][r] + xr[r * G5 + n] + bh[n];
          }
        }
      }
    }
    __syncthreads();

    // ---- phase 2: cumsoftmax on 32 rows (b in 0..7, gate group g in 1..4), wave-per-row
    {
      int b = wv & 7, g = 1 + (wv >> 3);
      float* row = sG + b * GP + g * HH;
      int base = lane * 10;
      bool actv = base < HH;            // lanes 0..29 own 10 elements each
      float v[10];
      float mx = -3.0e38f;
      if (actv) {
#pragma unroll
        for (int k = 0; k < 10; ++k) { v[k] = row[base + k]; mx = fmaxf(mx, v[k]); }
      }
      for (int off = 16; off; off >>= 1) mx = fmaxf(mx, __shfl_xor(mx, off, 32));
      float s = 0.f;
      if (actv) {
#pragma unroll
        for (int k = 0; k < 10; ++k) { v[k] = __expf(v[k] - mx); s += v[k]; }
      }
      float tot = s;
      for (int off = 16; off; off >>= 1) tot += __shfl_xor(tot, off, 32);
      float inv = 1.f / tot;
      float pre = s;                    // inclusive scan of chunk sums
      for (int d = 1; d < 32; d <<= 1) {
        float x = __shfl_up(pre, d, 32);
        if (lane >= d) pre += x;
      }
      pre -= s;                         // exclusive prefix
      if (actv) {
        bool oneMinus = (g == 1) || (g == 3);   // erase gates: 1 - cumsum
        float run = pre;
#pragma unroll
        for (int k = 0; k < 10; ++k) {
          run += v[k];
          float cum = run * inv;
          row[base + k] = oneMinus ? (1.f - cum) : cum;
        }
      }
    }
    __syncthreads();

    // ---- phase 3: pointwise combine -> registers, emit tanh outputs to global
    float cre[3], cner[3], shr[3];
    int cnt = 0;
    for (int e = tid; e < BB * HH; e += 1024) {
      int b = e / HH, j = e % HH;
      const float* gb = sG + b * GP;
      float cg  = tanhf(gb[j]);
      float egi = gb[HH + j],  rgi = gb[2 * HH + j];
      float egc = gb[3 * HH + j], rgc = gb[4 * HH + j];
      float cin = sC[b * 304 + j];
      float ovc = rgc * egc, upc = rgc - ovc, dnc = egc - ovc;
      float ovi = rgi * egi, upi = rgi - ovi, dni = egi - ovi;
      float sh = ovi * cin + ovc * cg;
      float cr = upi * cin + upc * cg + sh;
      float cn = dni * cin + dnc * cg + sh;
      size_t go = (size_t)(t * BB + b) * HP + j;
      hRe[go]    = (bf16)tanhf(cr);
      hNer[go]   = (bf16)tanhf(cn);
      hShare[go] = (bf16)tanhf(sh);
      cre[cnt] = cr; cner[cnt] = cn; shr[cnt] = sh; ++cnt;
    }
    __syncthreads();
    // write concat [c_re | c_ner | share] (each K-padded to 320) as bf16 over the gates region
    cnt = 0;
    for (int e = tid; e < BB * HH; e += 1024) {
      int b = e / HH, j = e % HH;
      sCC[b * 960 + j]            = (bf16)cre[cnt];
      sCC[b * 960 + HP + j]       = (bf16)cner[cnt];
      sCC[b * 960 + 2 * HP + j]   = (bf16)shr[cnt];
      ++cnt;
    }
    if (tid < 480) {   // zero the 3x20 pad columns per batch (region is rewritten every step)
      int b = tid / 60, r2 = tid % 60, sg = r2 / 20, p = HH + (r2 % 20);
      sCC[b * 960 + sg * HP + p] = (bf16)0.f;
    }
    __syncthreads();

    // ---- phase 4: c_out = cc @ Wt^T + bt ; h = tanh(c_out)
    if (wv < 19) {
      v8f acc = {};
      int n = wv * 16 + ml;
      int nc = (n < HH) ? n : (HH - 1);
      const bf16* Arow = sCC + (size_t)mc * 960;
      const bf16* Bp = WtBf + (size_t)nc * 960 + khalf * 16;
#pragma unroll
      for (int kc = 0; kc < 30; ++kc) {
        int k0 = kc * 32;
        v16bf af = load_afrag(Arow, k0, khalf);
        acc = wmma_bf16(af, load_bfrag(Bp + k0), acc);
      }
      if (lane < 16 && n < HH) {
#pragma unroll
        for (int r = 0; r < 8; ++r) {
          float cv = acc[r] + bt[n];
          sC[r * 304 + n] = cv;
          sH[r * HP + n]  = (bf16)tanhf(cv);
        }
      }
    }
    __syncthreads();
  }
}

// ---------------- fused pairwise head: one wave per (i,j,b) ----------------
// out[i,j,b,:] = sigmoid( W2 @ elu(LN(a[i,b]+b[j,b]+c[b])) + b2 ) * factor
__global__ void pairwise_kernel(const float* __restrict__ aP, const float* __restrict__ bP,
                                const float* __restrict__ cP, const float* __restrict__ gamma,
                                const float* __restrict__ beta, const float* __restrict__ W2,
                                const float* __restrict__ b2, const float* __restrict__ maskLB,
                                float* __restrict__ outP, int NTR, int useTri) {
  int wv = blockIdx.x * (blockDim.x >> 5) + (threadIdx.x >> 5);
  if (wv >= LL * LL * BB) return;
  int lane = threadIdx.x & 31;
  int i = wv / (LL * BB);
  int rem = wv % (LL * BB);
  int j = rem / BB, b = rem % BB;
  float factor = maskLB[i * BB + b] * maskLB[j * BB + b];
  if (useTri && j < i) factor = 0.f;
  float* out = outP + (size_t)wv * NTR;
  if (factor == 0.f) {
    for (int tt = lane; tt < NTR; tt += 32) out[tt] = 0.f;
    return;
  }
  const float* ar = aP + (size_t)(i * BB + b) * HH;
  const float* br = bP + (size_t)(j * BB + b) * HH;
  const float* cr = cP + (size_t)b * HH;
  float e[10];
  float s = 0.f, ss = 0.f;
  int cnt = 0;
  for (int k = lane; k < HH; k += 32) {
    float v = ar[k] + br[k] + cr[k];
    e[cnt++] = v; s += v; ss += v * v;
  }
  for (int off = 16; off; off >>= 1) {
    s  += __shfl_xor(s,  off, 32);
    ss += __shfl_xor(ss, off, 32);
  }
  const float invH = 1.f / (float)HH;
  float mean = s * invH;
  float var  = ss * invH - mean * mean;
  float rstd = rsqrtf(var + 1e-5f);
  cnt = 0;
  for (int k = lane; k < HH; k += 32) {
    float ln = (e[cnt] - mean) * rstd * gamma[k] + beta[k];
    e[cnt] = (ln > 0.f) ? ln : (__expf(ln) - 1.f);   // elu
    ++cnt;
  }
  for (int tt = 0; tt < NTR; ++tt) {
    const float* w = W2 + (size_t)tt * HH;
    float p = 0.f;
    cnt = 0;
    for (int k = lane; k < HH; k += 32) p += e[cnt++] * w[k];
    for (int off = 16; off; off >>= 1) p += __shfl_xor(p, off, 32);
    if (lane == 0) out[tt] = factor / (1.f + __expf(-(p + b2[tt])));
  }
}

// ---------------- host launcher ----------------
static inline size_t align256(size_t x) { return (x + 255) & ~(size_t)255; }

extern "C" void kernel_launch(void* const* d_in, const int* in_sizes, int n_in,
                              void* d_out, int out_size, void* d_ws, size_t ws_size,
                              hipStream_t stream) {
  (void)in_sizes; (void)n_in; (void)out_size; (void)ws_size;
  const float* x      = (const float*)d_in[0];
  const float* mask   = (const float*)d_in[1];
  const float* enc_Wi = (const float*)d_in[2];
  const float* enc_bi = (const float*)d_in[3];
  const float* enc_Wh = (const float*)d_in[4];
  const float* enc_bh = (const float*)d_in[5];
  const float* enc_Wt = (const float*)d_in[6];
  const float* enc_bt = (const float*)d_in[7];
  const float* ner_Wn = (const float*)d_in[8];
  const float* ner_bn = (const float*)d_in[9];
  const float* ner_W1 = (const float*)d_in[10];
  const float* ner_b1 = (const float*)d_in[11];
  const float* ner_g  = (const float*)d_in[12];
  const float* ner_be = (const float*)d_in[13];
  const float* ner_W2 = (const float*)d_in[14];
  const float* ner_b2 = (const float*)d_in[15];
  const float* re_Wr  = (const float*)d_in[16];
  const float* re_br  = (const float*)d_in[17];
  const float* re_W1  = (const float*)d_in[18];
  const float* re_b1  = (const float*)d_in[19];
  const float* re_g   = (const float*)d_in[20];
  const float* re_be  = (const float*)d_in[21];
  const float* re_W2  = (const float*)d_in[22];
  const float* re_b2  = (const float*)d_in[23];

  float* outNer = (float*)d_out;                              // [L,L,B,NT]
  float* outRe  = outNer + (size_t)LL * LL * BB * NTT;        // [L,L,B,NR]
  float* outReG = outRe  + (size_t)LL * LL * BB * NRR;        // [L,B,H]

  // workspace carve
  unsigned char* ws = (unsigned char*)d_ws;
  size_t off = 0;
  auto carve = [&](size_t bytes) { size_t o = off; off += align256(bytes); return o; };
  bf16*  xBf    = (bf16*)(ws + carve((size_t)MROWS * DD * 2));
  bf16*  WiBf   = (bf16*)(ws + carve((size_t)G5 * DD * 2));
  bf16*  WhBf   = (bf16*)(ws + carve((size_t)G5 * HP * 2));
  bf16*  WtBf   = (bf16*)(ws + carve((size_t)HH * 3 * HP * 2));
  bf16*  WnBf   = (bf16*)(ws + carve((size_t)HH * 2 * HP * 2));
  bf16*  WrBf   = (bf16*)(ws + carve((size_t)HH * 2 * HP * 2));
  bf16*  W1nBf  = (bf16*)(ws + carve((size_t)HH * 3 * HP * 2));
  bf16*  W1rBf  = (bf16*)(ws + carve((size_t)HH * 3 * HP * 2));
  float* xg     = (float*)(ws + carve((size_t)MROWS * G5 * 4));
  bf16*  hNerBf = (bf16*)(ws + carve((size_t)3 * MROWS * HP * 2));  // contiguous h{ner,re,share}
  bf16*  hReBf    = hNerBf + (size_t)MROWS * HP;
  bf16*  hShareBf = hReBf  + (size_t)MROWS * HP;
  float* hgPre  = (float*)(ws + carve((size_t)MROWS * HH * 4));
  bf16*  hgBf   = (bf16*)(ws + carve((size_t)BB * HP * 2));
  bf16*  rgBf   = (bf16*)(ws + carve((size_t)BB * HP * 2));
  float* aNer   = (float*)(ws + carve((size_t)MROWS * HH * 4));
  float* bNer   = (float*)(ws + carve((size_t)MROWS * HH * 4));
  float* aRe    = (float*)(ws + carve((size_t)MROWS * HH * 4));
  float* bRe    = (float*)(ws + carve((size_t)MROWS * HH * 4));
  float* cNer   = (float*)(ws + carve((size_t)BB * HH * 4));
  float* cRe    = (float*)(ws + carve((size_t)BB * HH * 4));

  const int T = 256;
  auto gblk = [&](long n) { return dim3((unsigned)((n + T - 1) / T)); };

  // --- weight/activation conversion to bf16 (K-padded per segment) ---
  cvt_pad_kernel<<<gblk((long)MROWS * DD), T, 0, stream>>>(x, xBf, MROWS, DD, DD, 1);
  cvt_pad_kernel<<<gblk((long)G5 * DD), T, 0, stream>>>(enc_Wi, WiBf, G5, DD, DD, 1);
  cvt_pad_kernel<<<gblk((long)G5 * HP), T, 0, stream>>>(enc_Wh, WhBf, G5, HH, HP, 1);
  cvt_pad_kernel<<<gblk((long)HH * 3 * HP), T, 0, stream>>>(enc_Wt, WtBf, HH, HH, HP, 3);
  cvt_pad_kernel<<<gblk((long)HH * 2 * HP), T, 0, stream>>>(ner_Wn, WnBf, HH, HH, HP, 2);
  cvt_pad_kernel<<<gblk((long)HH * 2 * HP), T, 0, stream>>>(re_Wr, WrBf, HH, HH, HP, 2);
  cvt_pad_kernel<<<gblk((long)HH * 3 * HP), T, 0, stream>>>(ner_W1, W1nBf, HH, HH, HP, 3);
  cvt_pad_kernel<<<gblk((long)HH * 3 * HP), T, 0, stream>>>(re_W1, W1rBf, HH, HH, HP, 3);
  zero_bf16_kernel<<<gblk((long)3 * MROWS * HP), T, 0, stream>>>(hNerBf, (long)3 * MROWS * HP);

  const int BIG = 1 << 30;
  auto gemmGrid = [&](int M, int N) {
    int Mt = (M + 15) / 16, Nt = (N + 15) / 16;
    int waves = Mt * ((Nt + NSUB - 1) / NSUB);
    return dim3((unsigned)((waves + 7) / 8));   // 8 waves / block
  };

  // --- phase A: xg = x @ Wi^T + bi  (M=800,N=1500,K=768) ---
  gemm_bf16_kernel<<<gemmGrid(MROWS, G5), 256, 0, stream>>>(
      xBf, DD, xBf, DD, BIG, WiBf, DD, enc_bi, xg, G5, MROWS, G5, DD, 0);

  // --- phase B: 100-step recurrence, single resident workgroup ---
  scan_kernel<<<1, 1024, 0, stream>>>(xg, enc_bh, enc_bt, WhBf, WtBf,
                                      hNerBf, hReBf, hShareBf);

  // --- phase C: head GEMMs ---
  // hg_pre = tanh([h_share|h_ner] @ Wn^T + bn)
  gemm_bf16_kernel<<<gemmGrid(MROWS, HH), 256, 0, stream>>>(
      hShareBf, HP, hNerBf, HP, HP, WnBf, 2 * HP, ner_bn, hgPre, HH, MROWS, HH, 2 * HP, 1);
  // re_global = tanh([h_share|h_re] @ Wr^T + br)  -> directly into d_out
  gemm_bf16_kernel<<<gemmGrid(MROWS, HH), 256, 0, stream>>>(
      hShareBf, HP, hReBf, HP, HP, WrBf, 2 * HP, re_br, outReG, HH, MROWS, HH, 2 * HP, 1);
  // hg / rg = max over L
  colmax_kernel<<<gblk(BB * HP), T, 0, stream>>>(hgPre, hgBf);
  colmax_kernel<<<gblk(BB * HP), T, 0, stream>>>(outReG, rgBf);
  // a/b projections (M=800,N=300,K=320) and c projections (M=8)
  gemm_bf16_kernel<<<gemmGrid(MROWS, HH), 256, 0, stream>>>(
      hNerBf, HP, hNerBf, HP, BIG, W1nBf, 3 * HP, nullptr, aNer, HH, MROWS, HH, HP, 0);
  gemm_bf16_kernel<<<gemmGrid(MROWS, HH), 256, 0, stream>>>(
      hNerBf, HP, hNerBf, HP, BIG, W1nBf + HP, 3 * HP, nullptr, bNer, HH, MROWS, HH, HP, 0);
  gemm_bf16_kernel<<<gemmGrid(MROWS, HH), 256, 0, stream>>>(
      hReBf, HP, hReBf, HP, BIG, W1rBf, 3 * HP, nullptr, aRe, HH, MROWS, HH, HP, 0);
  gemm_bf16_kernel<<<gemmGrid(MROWS, HH), 256, 0, stream>>>(
      hReBf, HP, hReBf, HP, BIG, W1rBf + HP, 3 * HP, nullptr, bRe, HH, MROWS, HH, HP, 0);
  gemm_bf16_kernel<<<gemmGrid(BB, HH), 256, 0, stream>>>(
      hgBf, HP, hgBf, HP, BIG, W1nBf + 2 * HP, 3 * HP, ner_b1, cNer, HH, BB, HH, HP, 0);
  gemm_bf16_kernel<<<gemmGrid(BB, HH), 256, 0, stream>>>(
      rgBf, HP, rgBf, HP, BIG, W1rBf + 2 * HP, 3 * HP, re_b1, cRe, HH, BB, HH, HP, 0);

  // --- phase D: fused pairwise heads (never materialize [L,L,B,H]) ---
  {
    long waves = (long)LL * LL * BB;
    dim3 grid((unsigned)((waves + 7) / 8));
    pairwise_kernel<<<grid, 256, 0, stream>>>(aNer, bNer, cNer, ner_g, ner_be,
                                              ner_W2, ner_b2, mask, outNer, NTT, 1);
    pairwise_kernel<<<grid, 256, 0, stream>>>(aRe, bRe, cRe, re_g, re_be,
                                              re_W2, re_b2, mask, outRe, NRR, 0);
  }
}
